// NNUE_18932215841063
// MI455X (gfx1250) — matrix operand order repaired and checked by reading, not compiled, not run on hardware
//
#include <hip/hip_runtime.h>
#include <hip/hip_bf16.h>

#define HALFKP   41024
#define FT_OUT   256
#define L1_OUT   32
#define L2_OUT   32
#define B_TOT    16384
#define K_ACT    30

#define BAGS_PER_WG 16
#define THREADS     256   // 8 waves (wave32)

// Feature row stride in LDS (halfs): 512 + 8 pad -> 1040 B/row -> 4-bank skew
// per row, de-conflicts the 16-lane A-fragment reads (1024 B would alias all
// rows onto the same banks since the bank period is 256 B).
#define FT_STRIDE 520

typedef __attribute__((ext_vector_type(16))) _Float16 v16h;
typedef __attribute__((ext_vector_type(8)))  float    v8f;
typedef __attribute__((ext_vector_type(2)))  _Float16 h2v;

__global__ __launch_bounds__(THREADS)
void nnue_fused_kernel(const int* __restrict__ w_idx,
                       const int* __restrict__ b_idx,
                       const int* __restrict__ stm,
                       const float* __restrict__ ft_w,
                       const float* __restrict__ ft_b,
                       const float* __restrict__ l1_w,
                       const float* __restrict__ l1_b,
                       const float* __restrict__ l2_w,
                       const float* __restrict__ l2_b,
                       const float* __restrict__ out_w,
                       const float* __restrict__ out_b,
                       float* __restrict__ out)
{
    __shared__ _Float16 ft_lds[BAGS_PER_WG * FT_STRIDE];   // [16][520] f16, ~16.6 KB
    __shared__ float    x1_lds[BAGS_PER_WG * L1_OUT];      // [16][32]
    __shared__ float    x2_lds[BAGS_PER_WG * L2_OUT];      // [16][32]

    const int tid     = threadIdx.x;
    const int wave    = tid >> 5;
    const int lane    = tid & 31;
    const int bagBase = blockIdx.x * BAGS_PER_WG;

    // -------- Phase 1: embedding-bag gather-sum (memory bound part) --------
    // Each wave owns a full 256-wide feature row: lane holds floats [8*lane, 8*lane+8).
    const float4 bias0 = *(const float4*)&ft_b[lane * 8];
    const float4 bias1 = *(const float4*)&ft_b[lane * 8 + 4];

    // 32 tasks = 16 bags x {white, black}; 8 waves -> 4 tasks each.
    for (int t = wave; t < 2 * BAGS_PER_WG; t += 8) {
        const int persp    = t >> 4;        // 0 = white, 1 = black
        const int bagLocal = t & 15;
        const int bag      = bagBase + bagLocal;
        const int* idxp    = (persp == 0 ? w_idx : b_idx) + bag * K_ACT;

        // Preload indices into lanes, broadcast via shfl (avoids 30 scalarised loads).
        int myidx = (lane < K_ACT) ? idxp[lane] : 0;

        float4 a0 = make_float4(0.f, 0.f, 0.f, 0.f);
        float4 a1 = make_float4(0.f, 0.f, 0.f, 0.f);

        for (int j = 0; j < K_ACT; ++j) {
            const int r = __shfl(myidx, j, 32);
            const float4* p = (const float4*)(ft_w + (size_t)r * FT_OUT + lane * 8);
            if (j + 4 < K_ACT) {
                const int rn = __shfl(myidx, j + 4, 32);
                __builtin_prefetch(ft_w + (size_t)rn * FT_OUT + lane * 8, 0, 0);
            }
            const float4 v0 = p[0];
            const float4 v1 = p[1];
            a0.x += v0.x; a0.y += v0.y; a0.z += v0.z; a0.w += v0.w;
            a1.x += v1.x; a1.y += v1.y; a1.z += v1.z; a1.w += v1.w;
        }

        float vals[8] = { a0.x + bias0.x, a0.y + bias0.y, a0.z + bias0.z, a0.w + bias0.w,
                          a1.x + bias1.x, a1.y + bias1.y, a1.z + bias1.z, a1.w + bias1.w };

        // stm==0 -> [white | black]; stm!=0 -> [black | white]
        const int stm_b = stm[bag];
        const int off   = ((persp ^ (stm_b != 0)) ? FT_OUT : 0);

        h2v* dst = (h2v*)&ft_lds[bagLocal * FT_STRIDE + off + lane * 8];
        #pragma unroll
        for (int k = 0; k < 4; ++k) {
            float lo = fminf(fmaxf(vals[2 * k],     0.f), 1.f);
            float hi = fminf(fmaxf(vals[2 * k + 1], 0.f), 1.f);
            h2v hp;
            hp.x = (_Float16)lo;
            hp.y = (_Float16)hi;
            dst[k] = hp;
        }
    }
    __syncthreads();

    // -------- Phase 2: L1 layer via WMMA: x1[16,32] = ft[16,512] @ l1_w.T --------
    // Waves 0 and 1 each produce one 16x16 N-tile; K = 512 in 16 chunks of 32.
    if (wave < 2) {
        const int nt = wave;            // N-tile
        const int lm = lane & 15;
        const int hs = lane >> 4;       // lane-half select
        const int n  = nt * 16 + lm;    // this lane's output column (C/D: N = lane%16)

        const float l1bias = l1_b[n];
        v8f c;
        #pragma unroll
        for (int r = 0; r < 8; ++r) c[r] = l1bias;   // fold bias into accumulator

        #pragma unroll
        for (int kc = 0; kc < 16; ++kc) {
            v16h a, bf;
            #pragma unroll
            for (int v = 0; v < 8; ++v) {
                // A 16x32 f16 layout: lanes0-15 VGPR v<4 -> K=2v,2v+1 ; lanes16-31 -> K+8;
                //                     VGPR v>=4 adds +16.
                const int kbase = ((v < 4) ? 0 : 16) + hs * 8 + 2 * (v & 3);
                const h2v pa = *(const h2v*)&ft_lds[lm * FT_STRIDE + kc * 32 + kbase];
                a[2 * v]     = pa.x;
                a[2 * v + 1] = pa.y;
                // B 32x16 f16: N across lanes (lane%16), K order same as A.
                const int kb = ((v < 4) ? 0 : 16) + hs * 8 + 2 * (v & 3);
                const float* wp = &l1_w[(size_t)n * 512 + kc * 32 + kb];
                bf[2 * v]     = (_Float16)wp[0];
                bf[2 * v + 1] = (_Float16)wp[1];
            }
            c = __builtin_amdgcn_wmma_f32_16x16x32_f16(
                    /*neg_a=*/false, a, /*neg_b=*/false, bf,
                    /*c_mod=*/(short)0, c, /*reuse_a=*/false, /*reuse_b=*/false);
        }

        // D layout: VGPR r holds M = r + 8*hs, N = lane%16. Clip and stash.
        #pragma unroll
        for (int r = 0; r < 8; ++r) {
            const int m = r + 8 * hs;
            x1_lds[m * L1_OUT + n] = fminf(fmaxf(c[r], 0.f), 1.f);
        }
    }
    __syncthreads();

    // -------- Phase 3: L2 layer (tiny: 16x32x32) --------
    for (int p = tid; p < BAGS_PER_WG * L2_OUT; p += THREADS) {
        const int bb = p >> 5;
        const int n  = p & 31;
        float s = l2_b[n];
        #pragma unroll
        for (int k = 0; k < L1_OUT; ++k)
            s += x1_lds[bb * L1_OUT + k] * l2_w[n * L1_OUT + k];
        x2_lds[bb * L2_OUT + n] = fminf(fmaxf(s, 0.f), 1.f);
    }
    __syncthreads();

    // -------- Phase 4: output layer (16 dots of length 32) --------
    if (tid < BAGS_PER_WG) {
        float s = out_b[0];
        #pragma unroll
        for (int k = 0; k < L2_OUT; ++k)
            s += x2_lds[tid * L2_OUT + k] * out_w[k];
        out[bagBase + tid] = s;
    }
}

extern "C" void kernel_launch(void* const* d_in, const int* in_sizes, int n_in,
                              void* d_out, int out_size, void* d_ws, size_t ws_size,
                              hipStream_t stream) {
    const int*   w_idx = (const int*)d_in[0];
    // d_in[1] = w_off: uniform bag*K, recomputed in-kernel (also avoids int64 dtype)
    const int*   b_idx = (const int*)d_in[2];
    // d_in[3] = b_off: unused (uniform)
    const int*   stm   = (const int*)d_in[4];
    const float* ft_w  = (const float*)d_in[5];
    const float* ft_b  = (const float*)d_in[6];
    const float* l1_w  = (const float*)d_in[7];
    const float* l1_b  = (const float*)d_in[8];
    const float* l2_w  = (const float*)d_in[9];
    const float* l2_b  = (const float*)d_in[10];
    const float* out_w = (const float*)d_in[11];
    const float* out_b = (const float*)d_in[12];
    float*       out   = (float*)d_out;

    dim3 grid(B_TOT / BAGS_PER_WG);   // 1024 workgroups
    dim3 block(THREADS);              // 8 wave32 waves
    nnue_fused_kernel<<<grid, block, 0, stream>>>(
        w_idx, b_idx, stm, ft_w, ft_b, l1_w, l1_b, l2_w, l2_b, out_w, out_b, out);
}
